// CrossEntropyLoss_69724499084016
// MI455X (gfx1250) — matrix operand order repaired and checked by reading, not compiled, not run on hardware
//
#include <hip/hip_runtime.h>
#include <hip/hip_bf16.h>

// ---------------- problem constants ----------------
#define D_DIM   2048
#define V_DIM   32000
#define B_DIM   4
#define S_DIM   2048
#define NROWS   8188          // B*(S-1)
#define NPAD    8192          // padded rows (multiple of 128)
#define VSPLIT  4
#define VPER    (V_DIM / VSPLIT)   // 8000
#define NC      32                 // vocab columns per chunk (2 WMMA N-tiles)
#define ROWS_PER_WG 128            // 8 waves * 16 rows
#define ROWBLOCKS  (NPAD / ROWS_PER_WG)  // 64
#define IGNORE_INDEX (-100)

// One staged W chunk in LDS: 32 rows x 2048 bf16 = 128 KB, stored 16B-transposed:
// element (row r, 16B-block z) lives at byte  z*512 + r*16  (conflict-free reads).
#define CHUNK_BYTES 131072
#define LDS_BYTES   (2 * CHUNK_BYTES)   // double buffered, 256 KB <= 320 KB/WGP

// ---------------- workspace layout (bytes) ----------------
#define WS_WB   ((size_t)0)                                   // bf16 W  [V_DIM * D_DIM]
#define WS_AB   (WS_WB + (size_t)V_DIM * D_DIM * 2)           // bf16 A  [NPAD * D_DIM]
#define WS_PM   (WS_AB + (size_t)NPAD * D_DIM * 2)            // float   [NPAD * VSPLIT]
#define WS_PS   (WS_PM + (size_t)NPAD * VSPLIT * 4)           // float   [NPAD * VSPLIT]
#define WS_GOLD (WS_PS + (size_t)NPAD * VSPLIT * 4)           // float   [NPAD]

// ---------------- CDNA5 vector types ----------------
typedef __attribute__((ext_vector_type(16))) __bf16       v16bf;
typedef __attribute__((ext_vector_type(8)))  float        v8f;
typedef __attribute__((ext_vector_type(4)))  unsigned int u32x4;
typedef __attribute__((ext_vector_type(8)))  int          i32x8;
typedef __attribute__((ext_vector_type(4)))  int          i32x4;
typedef __attribute__((ext_vector_type(4)))  __bf16       v4bf;

union Frag16 { v16bf bf; u32x4 q[2]; };

#if defined(__has_builtin)
#if __has_builtin(__builtin_amdgcn_tensor_load_to_lds) && __has_builtin(__builtin_amdgcn_s_wait_tensorcnt)
#define HAVE_TDM 1
#endif
#endif
#ifndef HAVE_TDM
#define HAVE_TDM 0
#endif

// ---------------- helpers ----------------
__device__ __forceinline__ unsigned lds_addr_of(const void* p) {
#if defined(__HIP_DEVICE_COMPILE__)
  return (unsigned)(unsigned long long)(__attribute__((address_space(3))) const char*)p;
#else
  (void)p; return 0;
#endif
}

#if HAVE_TDM
// TDM: DMA one W chunk (32 rows x 4096 B) into LDS with a 16B-granularity
// transpose, described as a 3-D tile: data_size=8B, tile=(x=2, y=32, z=256),
// tensor_dim0_stride = 512 (one W row in 8B units), tensor_dim1_stride = 2 (16B).
// LDS receives elements linearly in (x,y,z) order -> byte off = z*512 + y*16 + x*8.
__device__ __forceinline__ void tdm_load_chunk(unsigned ldsOff, const __bf16* gsrc) {
  unsigned long long ga = (unsigned long long)(uintptr_t)gsrc;
  u32x4 g0 = { 1u,                                   // count=1 valid descriptor
               ldsOff,                               // lds_addr [63:32]
               (unsigned)ga,                         // global_addr low
               (unsigned)(ga >> 32) | (2u << 30) };  // global_addr hi | type=2
  i32x8 g1 = { (int)(3u << 16),                      // data_size=3 (8 bytes)
               (int)(2u << 16),                      // tensor_dim0 = 2  (x extent)
               (int)(32u << 16),                     // tensor_dim1 = 32 (rows)
               (int)(2u << 16),                      // tile_dim0 = 2
               (int)(32u | (256u << 16)),            // tile_dim1=32, tile_dim2=256
               512,                                  // tensor_dim0_stride = 512 (8B units)
               (int)(2u << 16),                      // tensor_dim1_stride = 2  (16B)
               0 };
  i32x4 g2 = { 256, 0, 0, 0 };                       // tensor_dim2 = 256
  i32x4 g3 = { 0, 0, 0, 0 };
#if __clang_major__ >= 23
  i32x8 g4 = { 0, 0, 0, 0, 0, 0, 0, 0 };
  __builtin_amdgcn_tensor_load_to_lds(g0, g1, g2, g3, g4, 0);
#else
  __builtin_amdgcn_tensor_load_to_lds(g0, g1, g2, g3, 0);
#endif
}
#else
// Fallback staging with the identical 16B-transposed layout.
__device__ __forceinline__ void stage_chunk_fallback(char* dst, const __bf16* gsrc) {
  const int tid = threadIdx.x;
  const int r = tid >> 3;            // 32 rows
  const int z0 = tid & 7;            // 16B blocks, 256 per row
  for (int j = 0; j < 32; ++j) {
    const int z = z0 + 8 * j;
    u32x4 v = *reinterpret_cast<const u32x4*>(gsrc + (size_t)r * D_DIM + z * 8);
    *reinterpret_cast<u32x4*>(dst + z * 512 + r * 16) = v;
  }
}
#endif

// ---------------- fp32 -> bf16 conversion of shifted hidden rows ----------------
__global__ __launch_bounds__(256) void ce_cvtA(const float* __restrict__ h,
                                               __bf16* __restrict__ Ab) {
  const int row = blockIdx.x;
  const int tid = threadIdx.x;
  v4bf* dst = reinterpret_cast<v4bf*>(Ab + (size_t)row * D_DIM);
  if (row < NROWS) {
    const int b = row / 2047;
    const int s = row - b * 2047;
    const float4* src =
        reinterpret_cast<const float4*>(h + (size_t)(b * S_DIM + s) * D_DIM);
    for (int e = tid; e < D_DIM / 4; e += 256) {
      float4 f = src[e];
      v4bf o;
      o.x = (__bf16)f.x; o.y = (__bf16)f.y; o.z = (__bf16)f.z; o.w = (__bf16)f.w;
      dst[e] = o;
    }
  } else {
    v4bf z; z.x = (__bf16)0.0f; z.y = (__bf16)0.0f; z.z = (__bf16)0.0f; z.w = (__bf16)0.0f;
    for (int e = tid; e < D_DIM / 4; e += 256) dst[e] = z;
  }
}

// ---------------- fp32 -> bf16 conversion of lm_head weight ----------------
__global__ __launch_bounds__(256) void ce_cvtW(const float* __restrict__ w,
                                               __bf16* __restrict__ Wb) {
  const int row = blockIdx.x;
  const int tid = threadIdx.x;
  const float4* src = reinterpret_cast<const float4*>(w + (size_t)row * D_DIM);
  v4bf* dst = reinterpret_cast<v4bf*>(Wb + (size_t)row * D_DIM);
  for (int e = tid; e < D_DIM / 4; e += 256) {
    float4 f = src[e];
    v4bf o;
    o.x = (__bf16)f.x; o.y = (__bf16)f.y; o.z = (__bf16)f.z; o.w = (__bf16)f.w;
    dst[e] = o;
  }
}

// ---------------- fused GEMM + online logsumexp + gold gather ----------------
// grid.x: row blocks of 128 rows (8 waves * 16), grid.y: V split.
// W chunks are TDM-DMA'd into LDS (double buffered); A fragments stream from L2.
__global__ __launch_bounds__(256) void ce_main(const __bf16* __restrict__ Wb,
                                               const __bf16* __restrict__ Ab,
                                               const int*    __restrict__ labels,
                                               float* __restrict__ pm,
                                               float* __restrict__ ps,
                                               float* __restrict__ gold) {
  extern __shared__ char ldsRaw[];

  const int tid      = threadIdx.x;
  const int wave     = tid >> 5;
  const int lane     = tid & 31;
  const int laneHalf = lane >> 4;
  const int laneMod  = lane & 15;
  const int rowTile  = blockIdx.x * ROWS_PER_WG + wave * 16;
  const int vsplit   = blockIdx.y;
  const int vBase    = vsplit * VPER;
  const int nch      = VPER / NC;   // 250 chunks per split

  // labels for this lane's 8 C-tile row slots (row = rowTile + 8*laneHalf + i)
  int lab[8];
  #pragma unroll
  for (int i = 0; i < 8; ++i) {
    const int rowG = rowTile + 8 * laneHalf + i;
    int l = -1;
    if (rowG < NROWS) {
      const int b = rowG / 2047;
      const int s = rowG - b * 2047;
      l = labels[b * S_DIM + s + 1];     // causal shift
    }
    lab[i] = l;
  }

  float rmax[8], rsum[8];
  #pragma unroll
  for (int i = 0; i < 8; ++i) { rmax[i] = -3.0e38f; rsum[i] = 0.0f; }

  // A fragment (16x32 bf16): lane holds row M=laneMod; q0 = K k+8h+{0..7},
  // q1 = K k+16+8h+{0..7}.
  const __bf16* Aptr = Ab + (size_t)(rowTile + laneMod) * D_DIM + 8 * laneHalf;
  // B fragment from transposed LDS chunk: byte off(k) = k*64 + h*512 + r*16,
  // q1 at +1024; N-tile 1 rows are r+16 -> +256.
  const int bLaneOff = laneHalf * 512 + laneMod * 16;

#if HAVE_TDM
  const unsigned ldsBase = lds_addr_of(ldsRaw);
  if (wave == 0) {
    tdm_load_chunk(ldsBase, Wb + (size_t)vBase * D_DIM);
    __builtin_amdgcn_s_wait_tensorcnt(0);
  }
  __syncthreads();
#else
  stage_chunk_fallback(ldsRaw, Wb + (size_t)vBase * D_DIM);
  __syncthreads();
#endif

  for (int c = 0; c < nch; ++c) {
    const int v0 = vBase + c * NC;
    const char* buf = ldsRaw + (size_t)(c & 1) * CHUNK_BYTES;

#if HAVE_TDM
    // kick off the DMA for the next chunk into the other buffer
    if (wave == 0 && (c + 1) < nch)
      tdm_load_chunk(ldsBase + (unsigned)(((c + 1) & 1) * CHUNK_BYTES),
                     Wb + (size_t)(v0 + NC) * D_DIM);
#endif

    v8f acc0 = {};
    v8f acc1 = {};
    #pragma unroll 4
    for (int k = 0; k < D_DIM; k += 32) {
      Frag16 a, b0, b1;
      a.q[0] = *reinterpret_cast<const u32x4*>(Aptr + k);
      a.q[1] = *reinterpret_cast<const u32x4*>(Aptr + k + 16);
      const char* bp = buf + k * 64 + bLaneOff;
      b0.q[0] = *reinterpret_cast<const u32x4*>(bp);
      b0.q[1] = *reinterpret_cast<const u32x4*>(bp + 1024);
      b1.q[0] = *reinterpret_cast<const u32x4*>(bp + 256);
      b1.q[1] = *reinterpret_cast<const u32x4*>(bp + 256 + 1024);
      acc0 = __builtin_amdgcn_wmma_f32_16x16x32_bf16(
          false, a.bf, false, b0.bf, (short)0, acc0, false, false);
      acc1 = __builtin_amdgcn_wmma_f32_16x16x32_bf16(
          false, a.bf, false, b1.bf, (short)0, acc1, false, false);
    }

    // online logsumexp update + gold logit capture
    const int col0 = v0 + laneMod;
    const int col1 = col0 + 16;
    #pragma unroll
    for (int i = 0; i < 8; ++i) {
      const float x0 = acc0[i];
      const float x1 = acc1[i];
      const float m  = rmax[i];
      const float nm = fmaxf(m, fmaxf(x0, x1));
      rsum[i] = rsum[i] * __expf(m - nm) + __expf(x0 - nm) + __expf(x1 - nm);
      rmax[i] = nm;
      if (lab[i] == col0) gold[rowTile + 8 * laneHalf + i] = x0;
      if (lab[i] == col1) gold[rowTile + 8 * laneHalf + i] = x1;
    }

#if HAVE_TDM
    if (wave == 0 && (c + 1) < nch) __builtin_amdgcn_s_wait_tensorcnt(0);
#else
    if ((c + 1) < nch) {
      stage_chunk_fallback(ldsRaw + (size_t)(((c + 1) & 1)) * CHUNK_BYTES,
                           Wb + (size_t)(v0 + NC) * D_DIM);
    }
#endif
    __syncthreads();
  }

  // merge (max, sumexp) across the 16 column-lanes of each half-wave
  #pragma unroll
  for (int off = 1; off < 16; off <<= 1) {
    #pragma unroll
    for (int i = 0; i < 8; ++i) {
      const float om = __shfl_xor(rmax[i], off, 32);
      const float os = __shfl_xor(rsum[i], off, 32);
      const float m  = rmax[i];
      const float nm = fmaxf(m, om);
      rsum[i] = rsum[i] * __expf(m - nm) + os * __expf(om - nm);
      rmax[i] = nm;
    }
  }
  if (laneMod == 0) {
    #pragma unroll
    for (int i = 0; i < 8; ++i) {
      const int rowG = rowTile + 8 * laneHalf + i;
      pm[rowG * VSPLIT + vsplit] = rmax[i];
      ps[rowG * VSPLIT + vsplit] = rsum[i];
    }
  }
}

// ---------------- final reduction: combine V-splits, mask, mean ----------------
__global__ __launch_bounds__(256) void ce_reduce(const float* __restrict__ pm,
                                                 const float* __restrict__ ps,
                                                 const float* __restrict__ gold,
                                                 const int*   __restrict__ labels,
                                                 float* __restrict__ out) {
  __shared__ float sSum[256];
  __shared__ float sCnt[256];
  const int tid = threadIdx.x;
  float sum = 0.0f, cnt = 0.0f;
  for (int n = tid; n < NROWS; n += 256) {
    float M = -3.0e38f;
    #pragma unroll
    for (int sp = 0; sp < VSPLIT; ++sp) M = fmaxf(M, pm[n * VSPLIT + sp]);
    float S = 0.0f;
    #pragma unroll
    for (int sp = 0; sp < VSPLIT; ++sp)
      S += ps[n * VSPLIT + sp] * __expf(pm[n * VSPLIT + sp] - M);
    const float lse = M + __logf(S);
    const int b = n / 2047;
    const int s = n - b * 2047;
    const int lab = labels[b * S_DIM + s + 1];
    if (lab != IGNORE_INDEX) { sum += lse - gold[n]; cnt += 1.0f; }
  }
  sSum[tid] = sum; sCnt[tid] = cnt;
  __syncthreads();
  for (int st = 128; st > 0; st >>= 1) {
    if (tid < st) { sSum[tid] += sSum[tid + st]; sCnt[tid] += sCnt[tid + st]; }
    __syncthreads();
  }
  if (tid == 0) out[0] = sSum[0] / fmaxf(sCnt[0], 1.0f);
}

// ---------------- host-side launcher ----------------
extern "C" void kernel_launch(void* const* d_in, const int* in_sizes, int n_in,
                              void* d_out, int out_size, void* d_ws, size_t ws_size,
                              hipStream_t stream) {
  const float* h      = (const float*)d_in[0];   // [B,S,D] fp32
  const float* w      = (const float*)d_in[1];   // [V,D]   fp32
  const int*   labels = (const int*)d_in[2];     // [B,S]
  float* out = (float*)d_out;

  char* ws = (char*)d_ws;
  __bf16* Wb   = (__bf16*)(ws + WS_WB);
  __bf16* Ab   = (__bf16*)(ws + WS_AB);
  float*  pm   = (float*)(ws + WS_PM);
  float*  ps   = (float*)(ws + WS_PS);
  float*  gold = (float*)(ws + WS_GOLD);

  ce_cvtW<<<V_DIM, 256, 0, stream>>>(w, Wb);
  ce_cvtA<<<NPAD, 256, 0, stream>>>(h, Ab);
  ce_main<<<dim3(ROWBLOCKS, VSPLIT), 256, LDS_BYTES, stream>>>(Wb, Ab, labels,
                                                               pm, ps, gold);
  ce_reduce<<<1, 256, 0, stream>>>(pm, ps, gold, labels, out);
}
